// MultiLatentAttention_2138893714158
// MI455X (gfx1250) — matrix operand order repaired and checked by reading, not compiled, not run on hardware
//
#include <hip/hip_runtime.h>
#include <hip/hip_bf16.h>
#include <stdint.h>

typedef __bf16 bf16;
typedef __attribute__((ext_vector_type(16))) __bf16   v16bf;
typedef __attribute__((ext_vector_type(8)))  float    v8f;
typedef __attribute__((ext_vector_type(8)))  uint32_t v8u;
typedef __attribute__((ext_vector_type(4)))  uint32_t v4u;

__device__ __forceinline__ v8f zero_v8f() {
  v8f r;
#pragma unroll
  for (int i = 0; i < 8; ++i) r[i] = 0.0f;
  return r;
}

// ---- async global->LDS (CDNA5, tracked by ASYNCcnt) ----
// Per-lane: LDS[dsaddr .. +15] = MEM[gaddr .. +15]
__device__ __forceinline__ void async_load_b128(void* lds_ptr, const void* gptr) {
  uint32_t lds_off = (uint32_t)(uintptr_t)lds_ptr;  // LDS aperture: addr[31:0] is the LDS byte address
  uint64_t ga = (uint64_t)(uintptr_t)gptr;
  asm volatile("global_load_async_to_lds_b128 %0, %1, off"
               :: "v"(lds_off), "v"(ga)
               : "memory");
}
__device__ __forceinline__ void wait_asynccnt0() {
  asm volatile("s_wait_asynccnt 0x0" ::: "memory");
}

// ---- WMMA fragment loaders (CDNA5 ISA 7.12.2 layouts, wave32) ----
// A 16x32 bf16: lane m=L%16, half=L/16; VGPR v holds K = (v>=4?16:0)+8*half+2*(v&3)
// Per half, each group of 4 VGPRs is 16 contiguous bytes -> two ds_load_b128.
// Requires: stride % 8 == 0 and 16B-aligned base.
__device__ __forceinline__ v16bf frag_a_lds(const bf16* base, int stride) {
  const int lane = threadIdx.x & 31;
  const int m = lane & 15, h = lane >> 4;
  const uint16_t* p = (const uint16_t*)base + m * stride + (h << 3);
  v4u lo = *(const v4u*)(p);
  v4u hi = *(const v4u*)(p + 16);
  v8u u = __builtin_shufflevector(lo, hi, 0, 1, 2, 3, 4, 5, 6, 7);
  return __builtin_bit_cast(v16bf, u);
}

// B 32x16 bf16 staged transposed as Bt[n][k]: lane n=L%16, half=L/16; K = (v>=4?16:0)+8*half+2*(v&3)
// two runs of 8 contiguous bf16 per lane -> two ds_load_b128.
__device__ __forceinline__ v16bf frag_b_lds(const bf16* base, int stride) {
  const int lane = threadIdx.x & 31;
  const int n = lane & 15, h = lane >> 4;
  const uint16_t* p = (const uint16_t*)base + n * stride + (h << 3);
  v4u lo = *(const v4u*)(p);
  v4u hi = *(const v4u*)(p + 16);
  v8u u = __builtin_shufflevector(lo, hi, 0, 1, 2, 3, 4, 5, 6, 7);
  return __builtin_bit_cast(v16bf, u);
}

__device__ __forceinline__ v8f wmma_bf16(v16bf a, v16bf b, v8f c) {
  return __builtin_amdgcn_wmma_f32_16x16x32_bf16(false, a, false, b, (short)0, c, false, false);
}

// ---------------- fp32 -> bf16 convert ----------------
__global__ void f32_to_bf16_kernel(const float* __restrict__ in, bf16* __restrict__ out, int64_t n) {
  int64_t i = (int64_t)blockIdx.x * blockDim.x + threadIdx.x;
  int64_t stride = (int64_t)gridDim.x * blockDim.x;
  for (; i < n; i += stride) out[i] = (bf16)in[i];
}

// ---------------- generic bf16 GEMM: C = A @ B (+bias) ----------------
// Tile 128x128, kstep 32, 256 threads (8 waves), each wave 32x64 (2x4 WMMA tiles).
// Register-staged double buffering: one barrier per K-step, global-load waits
// sink past the WMMA block.
#define GT_M 128
#define GT_N 128
#define GT_K 32
#define LDA_S 40
#define LDB_S 40

template <bool OUTF32>
__global__ __launch_bounds__(256) void gemm_bf16_kernel(
    const bf16* __restrict__ A, int lda, int64_t aHead,
    const bf16* __restrict__ B, int ldb, int64_t bHead,
    const float* __restrict__ bias,
    void* __restrict__ Cout, int ldc, int64_t cHead,
    int Kdim) {
  __shared__ __align__(16) bf16 As[2][GT_M * LDA_S];
  __shared__ __align__(16) bf16 Bs[2][GT_N * LDB_S];

  const int tid = threadIdx.x;
  const int m0 = blockIdx.y * GT_M;
  const int n0 = blockIdx.x * GT_N;
  A += (int64_t)blockIdx.z * aHead + (int64_t)m0 * lda;
  B += (int64_t)blockIdx.z * bHead + n0;

  const int wid = tid >> 5;
  const int wm = (wid & 3) * 32;
  const int wn = (wid >> 2) * 64;

  v8f acc[2][4];
#pragma unroll
  for (int i = 0; i < 2; ++i)
#pragma unroll
    for (int j = 0; j < 4; ++j) acc[i][j] = zero_v8f();

  // per-thread staging registers: 2 x 16B for A-tile, 2 x 16B for B-tile
  auto load_tile = [&](int k0, v4u ra[2], v4u rb[2]) {
#pragma unroll
    for (int j = 0; j < 2; ++j) {
      int c = tid + j * 256;
      int r = c >> 2, ku = (c & 3) << 3;                 // A: 128 rows x 32k (4 chunks/row)
      ra[j] = *(const v4u*)(A + (int64_t)r * lda + k0 + ku);
      int kr = c >> 4, nu = (c & 15) << 3;               // B: 32k rows x 128n (16 chunks/row)
      rb[j] = *(const v4u*)(B + (int64_t)(k0 + kr) * ldb + nu);
    }
  };
  auto store_tile = [&](int buf, const v4u ra[2], const v4u rb[2]) {
#pragma unroll
    for (int j = 0; j < 2; ++j) {
      int c = tid + j * 256;
      int r = c >> 2, ku = (c & 3) << 3;
      *(v4u*)(&As[buf][r * LDA_S + ku]) = ra[j];
      int kr = c >> 4, nu = (c & 15) << 3;
#pragma unroll
      for (int e = 0; e < 4; ++e) {                      // transpose -> Bs[n][k]
        Bs[buf][(nu + 2 * e + 0) * LDB_S + kr] =
            __builtin_bit_cast(bf16, (uint16_t)(rb[j][e] & 0xFFFFu));
        Bs[buf][(nu + 2 * e + 1) * LDB_S + kr] =
            __builtin_bit_cast(bf16, (uint16_t)(rb[j][e] >> 16));
      }
    }
  };

  v4u ra[2], rb[2];
  load_tile(0, ra, rb);
  store_tile(0, ra, rb);
  int cur = 0;

  for (int k0 = 0; k0 < Kdim; k0 += GT_K) {
    __syncthreads();
    const bool hasNext = (k0 + GT_K < Kdim);
    if (hasNext) {
      __builtin_prefetch(A + (int64_t)(tid >> 1) * lda + k0 + 2 * GT_K, 0, 1);  // global_prefetch_b8
      load_tile(k0 + GT_K, ra, rb);
    }
    v16bf af[2], bfr[4];
#pragma unroll
    for (int i = 0; i < 2; ++i) af[i] = frag_a_lds(&As[cur][(wm + i * 16) * LDA_S], LDA_S);
#pragma unroll
    for (int j = 0; j < 4; ++j) bfr[j] = frag_b_lds(&Bs[cur][(wn + j * 16) * LDB_S], LDB_S);
#pragma unroll
    for (int i = 0; i < 2; ++i)
#pragma unroll
      for (int j = 0; j < 4; ++j) acc[i][j] = wmma_bf16(af[i], bfr[j], acc[i][j]);
    if (hasNext) store_tile(cur ^ 1, ra, rb);
    cur ^= 1;
  }

  // epilogue: C-frag layout lane n=L%16, row = v + 8*(L/16)
  const int lane = tid & 31;
  const int nin = lane & 15, h = lane >> 4;
#pragma unroll
  for (int i = 0; i < 2; ++i) {
#pragma unroll
    for (int j = 0; j < 4; ++j) {
      int col = n0 + wn + j * 16 + nin;
      float bv = bias ? bias[col] : 0.0f;
#pragma unroll
      for (int v = 0; v < 8; ++v) {
        int row = m0 + wm + i * 16 + h * 8 + v;
        float val = acc[i][j][v] + bv;
        int64_t idx = (int64_t)blockIdx.z * cHead + (int64_t)row * ldc + col;
        if (OUTF32) ((float*)Cout)[idx] = val;
        else        ((bf16*)Cout)[idx] = (bf16)val;
      }
    }
  }
}

// ---------------- causal flash attention (bf16 WMMA, online softmax) ----------------
// grid (S/64, H, B), block 128 threads (4 waves x 16 q-rows). Key tiles of 64.
// Q and K tiles staged with async global->LDS DMA (ASYNCcnt); V staged through
// VGPRs (needs transpose).
#define FA_QT 64
#define FA_KT 64
#define FA_DK 128
#define FA_DV 128
#define QS_S 136
#define KS_S 136
#define VT_S 72
#define P_S 72

__global__ __launch_bounds__(128) void mla_flash_kernel(
    const bf16* __restrict__ Q, const bf16* __restrict__ Kk,
    const bf16* __restrict__ Vv, bf16* __restrict__ O, int S, int H) {
  __shared__ __align__(16) bf16 Qs[FA_QT * QS_S];
  __shared__ __align__(16) bf16 Ks[FA_KT * KS_S];
  __shared__ __align__(16) bf16 Vts[FA_DV * VT_S];
  __shared__ __align__(16) bf16 Ps[4][16 * P_S];

  const int tid = threadIdx.x;
  const int lane = tid & 31;
  const int w = tid >> 5;
  const int nin = lane & 15, half = lane >> 4;

  const int qb = blockIdx.x * FA_QT;
  const int h = blockIdx.y;
  const int b = blockIdx.z;
  const int ld = H * FA_DK;
  const int64_t rowOff = (int64_t)b * S;

  // stage Q tile [64][128] via async DMA to LDS
  const bf16* Qg = Q + (rowOff + qb) * (int64_t)ld + h * FA_DK;
#pragma unroll 4
  for (int c = tid; c < FA_QT * FA_DK / 8; c += 128) {
    int r = c >> 4, ku = (c & 15) << 3;
    async_load_b128(&Qs[r * QS_S + ku], Qg + (int64_t)r * ld + ku);
  }

  v8f Oacc[8];
#pragma unroll
  for (int j = 0; j < 8; ++j) Oacc[j] = zero_v8f();
  float mst[8], lst[8];
#pragma unroll
  for (int v = 0; v < 8; ++v) { mst[v] = -1e30f; lst[v] = 0.0f; }

  const int ktEnd = blockIdx.x;              // causal: key tiles 0..qtile
  const float scale = 0.08838834764831845f;  // 1/sqrt(128)

  for (int kt = 0; kt <= ktEnd; ++kt) {
    const int k0 = kt * FA_KT;
    __syncthreads();
    // stage K tile [64][128] row-major (serves as Bt for QK^T) via async DMA,
    // and V transposed Vt[d][key] through VGPRs
    const bf16* Kg = Kk + (rowOff + k0) * (int64_t)ld + h * FA_DK;
    const bf16* Vg = Vv + (rowOff + k0) * (int64_t)ld + h * FA_DV;
#pragma unroll 4
    for (int c = tid; c < FA_KT * FA_DK / 8; c += 128) {
      int r = c >> 4, ku = (c & 15) << 3;
      async_load_b128(&Ks[r * KS_S + ku], Kg + (int64_t)r * ld + ku);
      v4u dv = *(const v4u*)(Vg + (int64_t)r * ld + ku);
#pragma unroll
      for (int e = 0; e < 4; ++e) {
        Vts[(ku + 2 * e + 0) * VT_S + r] = __builtin_bit_cast(bf16, (uint16_t)(dv[e] & 0xFFFFu));
        Vts[(ku + 2 * e + 1) * VT_S + r] = __builtin_bit_cast(bf16, (uint16_t)(dv[e] >> 16));
      }
    }
    wait_asynccnt0();  // drain async DMA before the consuming barrier
    __syncthreads();

    // S = Q K^T : wave's 16 q-rows x 64 keys
    v8f sfr[4];
#pragma unroll
    for (int nf = 0; nf < 4; ++nf) sfr[nf] = zero_v8f();
#pragma unroll
    for (int kk = 0; kk < FA_DK; kk += 32) {
      v16bf a = frag_a_lds(&Qs[(w * 16) * QS_S + kk], QS_S);
#pragma unroll
      for (int nf = 0; nf < 4; ++nf) {
        v16bf bb = frag_b_lds(&Ks[(nf * 16) * KS_S + kk], KS_S);
        sfr[nf] = wmma_bf16(a, bb, sfr[nf]);
      }
    }

    // scale + causal mask (only the diagonal tile needs it)
    const int qrow0 = qb + w * 16 + half * 8;
    const bool needMask = (kt == ktEnd);
#pragma unroll
    for (int nf = 0; nf < 4; ++nf) {
      int kcol = k0 + nf * 16 + nin;
#pragma unroll
      for (int v = 0; v < 8; ++v) {
        float s = sfr[nf][v] * scale;
        if (needMask && kcol > qrow0 + v) s = -1e30f;
        sfr[nf][v] = s;
      }
    }

    // row max (reduce across 4 n-frags, then across 16 lanes of the half)
    float mnew[8];
#pragma unroll
    for (int v = 0; v < 8; ++v) {
      float t = fmaxf(fmaxf(sfr[0][v], sfr[1][v]), fmaxf(sfr[2][v], sfr[3][v]));
#pragma unroll
      for (int off = 8; off >= 1; off >>= 1) t = fmaxf(t, __shfl_xor(t, off, 16));
      mnew[v] = fmaxf(mst[v], t);
    }

    // exp, row sums, rescale O
    float rsum[8];
#pragma unroll
    for (int v = 0; v < 8; ++v) {
      float corr = __expf(mst[v] - mnew[v]);
      lst[v] *= corr;
#pragma unroll
      for (int j = 0; j < 8; ++j) Oacc[j][v] *= corr;
      mst[v] = mnew[v];
      float sum = 0.0f;
#pragma unroll
      for (int nf = 0; nf < 4; ++nf) {
        float p = __expf(sfr[nf][v] - mnew[v]);
        sfr[nf][v] = p;
        sum += p;
      }
      rsum[v] = sum;
    }
#pragma unroll
    for (int v = 0; v < 8; ++v) {
#pragma unroll
      for (int off = 8; off >= 1; off >>= 1) rsum[v] += __shfl_xor(rsum[v], off, 16);
      lst[v] += rsum[v];
    }

    // write P (bf16) into per-wave LDS tile [16 q][64 keys]
    bf16* Pw = Ps[w];
#pragma unroll
    for (int nf = 0; nf < 4; ++nf)
#pragma unroll
      for (int v = 0; v < 8; ++v)
        Pw[(half * 8 + v) * P_S + nf * 16 + nin] = (bf16)sfr[nf][v];
    __syncthreads();

    // O += P @ V (Vt[d][key] is Bt layout)
#pragma unroll
    for (int kk = 0; kk < FA_KT; kk += 32) {
      v16bf a = frag_a_lds(&Pw[kk], P_S);
#pragma unroll
      for (int nf = 0; nf < 8; ++nf) {
        v16bf bb = frag_b_lds(&Vts[(nf * 16) * VT_S + kk], VT_S);
        Oacc[nf] = wmma_bf16(a, bb, Oacc[nf]);
      }
    }
  }

  // finalize: O /= l, store bf16 [B*S, H*DV]
  const int64_t ldo = (int64_t)H * FA_DV;
#pragma unroll
  for (int v = 0; v < 8; ++v) {
    float inv = 1.0f / lst[v];
    int64_t row = rowOff + qb + w * 16 + half * 8 + v;
#pragma unroll
    for (int nf = 0; nf < 8; ++nf)
      O[row * ldo + h * FA_DV + nf * 16 + nin] = (bf16)(Oacc[nf][v] * inv);
  }
}

// ---------------- host launcher ----------------
extern "C" void kernel_launch(void* const* d_in, const int* in_sizes, int n_in,
                              void* d_out, int out_size, void* d_ws, size_t ws_size,
                              hipStream_t stream) {
  (void)in_sizes; (void)n_in; (void)out_size; (void)ws_size;
  constexpr int B = 2, S = 2048, D = 2048, H = 16, DK = 128, DV = 128, LDm = 64;
  constexpr int M = B * S;    // 4096
  constexpr int NQ = H * DK;  // 2048
  constexpr int NL = H * LDm; // 1024

  const float* queries = (const float*)d_in[0];
  const float* keys    = (const float*)d_in[1];
  const float* values  = (const float*)d_in[2];
  const float* Wq      = (const float*)d_in[3];
  const float* bq      = (const float*)d_in[4];
  const float* Wlk     = (const float*)d_in[5];
  const float* blk_    = (const float*)d_in[6];
  const float* Wlv     = (const float*)d_in[7];
  const float* blv     = (const float*)d_in[8];
  const float* Wkr     = (const float*)d_in[9];
  const float* Wvr     = (const float*)d_in[10];
  const float* Wout    = (const float*)d_in[11];
  const float* bout    = (const float*)d_in[12];

  char* ws = (char*)d_ws;
  size_t off = 0;
  auto take = [&](size_t elems) -> bf16* {
    bf16* p = (bf16*)(ws + off);
    off += (elems * sizeof(bf16) + 255) & ~(size_t)255;
    return p;
  };
  bf16* qb    = take((size_t)M * D);
  bf16* kb    = take((size_t)M * D);
  bf16* vb    = take((size_t)M * D);
  bf16* wqb   = take((size_t)D * NQ);
  bf16* wlkb  = take((size_t)D * NL);
  bf16* wlvb  = take((size_t)D * NL);
  bf16* wkrb  = take((size_t)LDm * DK);
  bf16* wvrb  = take((size_t)LDm * DV);
  bf16* woutb = take((size_t)NQ * D);
  bf16* Qb    = take((size_t)M * NQ);
  bf16* LKb   = take((size_t)M * NL);
  bf16* LVb   = take((size_t)M * NL);
  bf16* Kb    = take((size_t)M * NQ);
  bf16* Vb    = take((size_t)M * NQ);
  bf16* AOb   = take((size_t)M * NQ);

  auto conv = [&](const float* src, bf16* dst, int64_t n) {
    int grid = (int)((n + 255) / 256);
    if (grid > 4096) grid = 4096;
    f32_to_bf16_kernel<<<grid, 256, 0, stream>>>(src, dst, n);
  };
  conv(queries, qb, (int64_t)M * D);
  conv(keys,    kb, (int64_t)M * D);
  conv(values,  vb, (int64_t)M * D);
  conv(Wq,   wqb,   (int64_t)D * NQ);
  conv(Wlk,  wlkb,  (int64_t)D * NL);
  conv(Wlv,  wlvb,  (int64_t)D * NL);
  conv(Wkr,  wkrb,  (int64_t)LDm * DK);
  conv(Wvr,  wvrb,  (int64_t)LDm * DV);
  conv(Wout, woutb, (int64_t)NQ * D);

  dim3 tpb(256);
  // Q / latent-K / latent-V projections
  gemm_bf16_kernel<false><<<dim3(NQ / 128, M / 128, 1), tpb, 0, stream>>>(
      qb, D, 0, wqb, NQ, 0, bq, Qb, NQ, 0, D);
  gemm_bf16_kernel<false><<<dim3(NL / 128, M / 128, 1), tpb, 0, stream>>>(
      kb, D, 0, wlkb, NL, 0, blk_, LKb, NL, 0, D);
  gemm_bf16_kernel<false><<<dim3(NL / 128, M / 128, 1), tpb, 0, stream>>>(
      vb, D, 0, wlvb, NL, 0, blv, LVb, NL, 0, D);
  // per-head K/V reconstruction from latents (gridDim.z = heads)
  gemm_bf16_kernel<false><<<dim3(1, M / 128, H), tpb, 0, stream>>>(
      LKb, NL, LDm, wkrb, DK, 0, nullptr, Kb, NQ, DK, LDm);
  gemm_bf16_kernel<false><<<dim3(1, M / 128, H), tpb, 0, stream>>>(
      LVb, NL, LDm, wvrb, DV, 0, nullptr, Vb, NQ, DV, LDm);
  // causal flash attention
  mla_flash_kernel<<<dim3(S / 64, H, B), dim3(128), 0, stream>>>(Qb, Kb, Vb, AOb, S, H);
  // output projection (fp32 epilogue + bias straight into d_out)
  gemm_bf16_kernel<true><<<dim3(D / 128, M / 128, 1), tpb, 0, stream>>>(
      AOb, NQ, 0, woutb, D, 0, bout, d_out, D, 0, NQ);
}
